// LanguageModel_10290741641531
// MI455X (gfx1250) — compile-verified, matrix operand order checked by reading, not run on hardware
//
#include <hip/hip_runtime.h>

typedef __bf16 bf16_t;
typedef __attribute__((ext_vector_type(16))) __bf16 v16bf;
typedef __attribute__((ext_vector_type(8)))  float  v8f;

#define Bv 64
#define Sv 512
#define Ev 128
#define Hv 1024
#define Ov 128

#define KT_H 32   // 1024/32 K-tiles
#define KT_E 4    // 128/32  K-tiles

// ---- packed-layout index helpers (ISA 7.12.2, 16-bit A 16x32 / B 32x16, wave32) ----
// Within a 16x32 tile, lane/element position of (row m, k):
//   lane = (m&15) | (((k>>3)&1)<<4)
//   slot = ((k>>4)<<2) | ((k&7)>>1), elem = slot*2 + (k&1)
// Tiles stored as [..tile..][32 lanes][16 bf16] so each lane reads 32 contiguous bytes.
__device__ __host__ __forceinline__ int packA_off(int m, int k, int ktiles) {
  int mt = m >> 4, kt = k >> 5, kk = k & 31;
  int lane = (m & 15) | (((kk >> 3) & 1) << 4);
  int slot = ((kk >> 4) << 2) | ((kk & 7) >> 1);
  return (((mt * ktiles + kt) * 32 + lane) << 4) | ((slot << 1) | (kk & 1));
}
__device__ __host__ __forceinline__ int packB_off(int k, int n, int ktiles) {
  int nt = n >> 4, kt = k >> 5, kk = k & 31;
  int lane = (n & 15) | (((kk >> 3) & 1) << 4);
  int slot = ((kk >> 4) << 2) | ((kk & 7) >> 1);
  return (((nt * ktiles + kt) * 32 + lane) << 4) | ((slot << 1) | (kk & 1));
}

__device__ __forceinline__ float sigmoidf_(float x) { return 1.f / (1.f + __expf(-x)); }

__device__ __forceinline__ v8f wmma_bf16(v16bf a, v16bf b, v8f c) {
  return __builtin_amdgcn_wmma_f32_16x16x32_bf16(false, a, false, b, (short)0, c,
                                                 false, false);
}

// ---- weight packer: fp32 [K,N] row-major -> bf16 B-tile-packed ----
__global__ void pack_b_kernel(const float* __restrict__ W, bf16_t* __restrict__ Wp,
                              int K, int N) {
  int idx = blockIdx.x * blockDim.x + threadIdx.x;
  if (idx >= K * N) return;
  int k = idx / N, n = idx - k * N;
  Wp[packB_off(k, n, K >> 5)] = (bf16_t)W[idx];
}

// ---- embedding gather + bf16 convert into per-step A-tile-packed X ----
// Xgru layout: [S][4 mtiles(B=64)][4 ktiles(E=128)][32][16] bf16, M = batch index b.
__global__ void embed_pack_kernel(const int* __restrict__ inputs,
                                  const float* __restrict__ emb,
                                  bf16_t* __restrict__ Xgru) {
  int idx = blockIdx.x * blockDim.x + threadIdx.x;   // over B*S*E
  if (idx >= Bv * Sv * Ev) return;
  int e  = idx & (Ev - 1);
  int bs = idx >> 7;          // b*S + s (inputs is [B,S] row-major)
  int b  = bs >> 9;
  int s  = bs & (Sv - 1);
  int tok = inputs[bs];
  Xgru[(size_t)s * (Bv * Ev) + packA_off(b, e, KT_E)] = (bf16_t)emb[tok * Ev + e];
}

// ---- GRU scan: 1 workgroup, h double-buffered in LDS, bf16 WMMA + f32 acc ----
__global__ void __launch_bounds__(512, 1)
gru_kernel(const bf16_t* __restrict__ Xgru,
           const bf16_t* __restrict__ WhrP, const bf16_t* __restrict__ WhzP,
           const bf16_t* __restrict__ WhnP,
           const bf16_t* __restrict__ WirP, const bf16_t* __restrict__ WizP,
           const bf16_t* __restrict__ WinP,
           const float* __restrict__ carry,
           const float* __restrict__ bir, const float* __restrict__ biz,
           const float* __restrict__ bin_, const float* __restrict__ bhn,
           bf16_t* __restrict__ Y, float* __restrict__ carry_out) {
  extern __shared__ char smem[];
  bf16_t* hbuf0 = (bf16_t*)smem;              // 64x1024 packed-A, 128 KB
  bf16_t* hbuf1 = hbuf0 + Bv * Hv;            // back buffer, 128 KB

  const int tid  = threadIdx.x;
  const int lane = tid & 31;
  const int wave = tid >> 5;                  // 0..15

  // h0 <- carry (fp32 -> bf16, packed-A layout)
  for (int i = tid; i < Bv * Hv; i += blockDim.x) {
    int m = i >> 10, k = i & (Hv - 1);
    hbuf0[packA_off(m, k, KT_H)] = (bf16_t)carry[i];
  }
  __syncthreads();

  for (int s = 0; s < Sv; ++s) {
    const bf16_t* hsrc = (s & 1) ? hbuf1 : hbuf0;
    bf16_t*       hdst = (s & 1) ? hbuf0 : hbuf1;
    const bf16_t* xbase = Xgru + (size_t)s * (Bv * Ev);

    // 256 output tiles (4 mtiles x 64 ntiles) over 16 waves -> 16 per wave
    for (int j = 0; j < 16; ++j) {
      int t  = wave * 16 + j;
      int mt = t >> 6;          // 0..3
      int nt = t & 63;          // 0..63
      v8f accr = {}, accz = {}, acchn = {}, accxn = {};

      // recurrent part: h @ Wh{r,z,n}, K = 1024
      #pragma unroll 2
      for (int kt = 0; kt < KT_H; ++kt) {
        v16bf a  = *(const v16bf*)(hsrc + (((mt * KT_H + kt) * 32 + lane) << 4));
        int   bb = ((nt * KT_H + kt) * 32 + lane) << 4;
        v16bf br = *(const v16bf*)(WhrP + bb);
        v16bf bz = *(const v16bf*)(WhzP + bb);
        v16bf bn = *(const v16bf*)(WhnP + bb);
        accr  = wmma_bf16(a, br, accr);
        accz  = wmma_bf16(a, bz, accz);
        acchn = wmma_bf16(a, bn, acchn);
      }
      // input part: x @ Wi{r,z,n}, K = 128
      #pragma unroll
      for (int kt = 0; kt < KT_E; ++kt) {
        v16bf a  = *(const v16bf*)(xbase + (((mt * KT_E + kt) * 32 + lane) << 4));
        int   bb = ((nt * KT_E + kt) * 32 + lane) << 4;
        v16bf br = *(const v16bf*)(WirP + bb);
        v16bf bz = *(const v16bf*)(WizP + bb);
        v16bf bn = *(const v16bf*)(WinP + bb);
        accr  = wmma_bf16(a, br, accr);
        accz  = wmma_bf16(a, bz, accz);
        accxn = wmma_bf16(a, bn, accxn);
      }

      // elementwise GRU update in C-matrix layout:
      // lane: N = lane&15, VGPR r: M = r + ((lane>>4)<<3)
      int   ncol = (nt << 4) + (lane & 15);
      float b_r  = bir[ncol], b_z = biz[ncol], b_in = bin_[ncol], b_hn = bhn[ncol];
      int   mbase = (mt << 4) + ((lane >> 4) << 3);
      #pragma unroll
      for (int rI = 0; rI < 8; ++rI) {
        int   m    = mbase + rI;                        // batch index 0..63
        float hold = (float)hsrc[packA_off(m, ncol, KT_H)];
        float rg   = sigmoidf_(accr[rI] + b_r);
        float zg   = sigmoidf_(accz[rI] + b_z);
        float ng   = tanhf(accxn[rI] + b_in + rg * (acchn[rI] + b_hn));
        float hn   = (1.f - zg) * ng + zg * hold;
        bf16_t hb  = (bf16_t)hn;
        hdst[packA_off(m, ncol, KT_H)] = hb;            // LDS back buffer
        // Y is packed-A over global rows m_g = b*S + s (matches reference reshape)
        Y[packA_off(m * Sv + s, ncol, KT_H)] = hb;
        if (s == Sv - 1) carry_out[m * Hv + ncol] = hn; // full-precision carry
      }
    }
    __syncthreads();  // all reads of hsrc done before it becomes hdst next step
  }
}

// ---- fused dense head: hidden = Y@Wh+bh (LDS-resident), logits = hidden@Wo+bo ----
__global__ void __launch_bounds__(256, 1)
dense_kernel(const bf16_t* __restrict__ Y, const bf16_t* __restrict__ WhP,
             const bf16_t* __restrict__ WoP, const float* __restrict__ bh,
             const float* __restrict__ bo, float* __restrict__ logits) {
  extern __shared__ char smem[];
  bf16_t* hid = (bf16_t*)smem;                // 64x1024 packed-A, 128 KB

  const int tid  = threadIdx.x;
  const int lane = tid & 31;
  const int wave = tid >> 5;                  // 0..7
  const int wg   = blockIdx.x;                // 0..511 -> rows [wg*64, wg*64+64)

  // dense1: 256 output tiles (4 local mtiles x 64 ntiles) / 8 waves = 32 each
  for (int j = 0; j < 32; ++j) {
    int t   = wave * 32 + j;
    int mtl = t >> 6;
    int nt  = t & 63;
    v8f acc = {};
    #pragma unroll 2
    for (int kt = 0; kt < 32; ++kt) {
      size_t ao = ((size_t)((wg * 4 + mtl) * 32 + kt) * 32 + lane) << 4;
      v16bf a = *(const v16bf*)(Y + ao);
      v16bf b = *(const v16bf*)(WhP + (((nt * 32 + kt) * 32 + lane) << 4));
      acc = wmma_bf16(a, b, acc);
    }
    int   ncol  = (nt << 4) + (lane & 15);
    float bias  = bh[ncol];
    int   mbase = (mtl << 4) + ((lane >> 4) << 3);
    #pragma unroll
    for (int rI = 0; rI < 8; ++rI)
      hid[packA_off(mbase + rI, ncol, 32)] = (bf16_t)(acc[rI] + bias);
  }
  __syncthreads();

  // dense2: 32 output tiles (4 mtiles x 8 ntiles) / 8 waves = 4 each
  for (int j = 0; j < 4; ++j) {
    int t   = wave * 4 + j;
    int mtl = t >> 3;
    int nt  = t & 7;
    v8f acc = {};
    #pragma unroll 2
    for (int kt = 0; kt < 32; ++kt) {
      v16bf a = *(const v16bf*)(hid + (((mtl * 32 + kt) * 32 + lane) << 4));
      v16bf b = *(const v16bf*)(WoP + (((nt * 32 + kt) * 32 + lane) << 4));
      acc = wmma_bf16(a, b, acc);
    }
    int   ncol  = (nt << 4) + (lane & 15);
    float bias  = bo[ncol];
    int   mbase = (mtl << 4) + ((lane >> 4) << 3);
    #pragma unroll
    for (int rI = 0; rI < 8; ++rI) {
      size_t mg = (size_t)wg * 64 + (mbase + rI);
      logits[mg * Ov + ncol] = acc[rI] + bias;
    }
  }
}

static inline size_t align_up(size_t x, size_t a) { return (x + a - 1) & ~(a - 1); }

extern "C" void kernel_launch(void* const* d_in, const int* in_sizes, int n_in,
                              void* d_out, int out_size, void* d_ws, size_t ws_size,
                              hipStream_t stream) {
  const int*   inputs = (const int*)  d_in[0];
  const float* carry  = (const float*)d_in[1];
  const float* emb    = (const float*)d_in[2];
  const float* Wir  = (const float*)d_in[3];  const float* bir  = (const float*)d_in[4];
  const float* Whr  = (const float*)d_in[5];
  const float* Wiz  = (const float*)d_in[6];  const float* biz  = (const float*)d_in[7];
  const float* Whz  = (const float*)d_in[8];
  const float* Win  = (const float*)d_in[9];  const float* bin_ = (const float*)d_in[10];
  const float* Whn  = (const float*)d_in[11]; const float* bhn  = (const float*)d_in[12];
  const float* Wh   = (const float*)d_in[13]; const float* bh   = (const float*)d_in[14];
  const float* Wo   = (const float*)d_in[15]; const float* bo   = (const float*)d_in[16];

  float* out       = (float*)d_out;
  float* carry_out = out;                 // [64,1024]
  float* logits    = out + Bv * Hv;       // [32768,128]

  // workspace partition (bf16 packed buffers), all 256B-aligned
  char* p = (char*)d_ws;
  auto take = [&](size_t bytes) { char* r = p; p += align_up(bytes, 256); return r; };
  bf16_t* Xgru = (bf16_t*)take((size_t)Sv * Bv * Ev * 2);   // 8.4 MB
  bf16_t* WirP = (bf16_t*)take((size_t)Ev * Hv * 2);
  bf16_t* WizP = (bf16_t*)take((size_t)Ev * Hv * 2);
  bf16_t* WinP = (bf16_t*)take((size_t)Ev * Hv * 2);
  bf16_t* WhrP = (bf16_t*)take((size_t)Hv * Hv * 2);
  bf16_t* WhzP = (bf16_t*)take((size_t)Hv * Hv * 2);
  bf16_t* WhnP = (bf16_t*)take((size_t)Hv * Hv * 2);
  bf16_t* WhP  = (bf16_t*)take((size_t)Hv * Hv * 2);
  bf16_t* WoP  = (bf16_t*)take((size_t)Hv * Ov * 2);
  bf16_t* Y    = (bf16_t*)take((size_t)Bv * Sv * Hv * 2);   // 67 MB

  const int T = 256;
  // pack weights into WMMA B-tile layout (one-shot, trivial cost)
  pack_b_kernel<<<(Ev * Hv + T - 1) / T, T, 0, stream>>>(Wir, WirP, Ev, Hv);
  pack_b_kernel<<<(Ev * Hv + T - 1) / T, T, 0, stream>>>(Wiz, WizP, Ev, Hv);
  pack_b_kernel<<<(Ev * Hv + T - 1) / T, T, 0, stream>>>(Win, WinP, Ev, Hv);
  pack_b_kernel<<<(Hv * Hv + T - 1) / T, T, 0, stream>>>(Whr, WhrP, Hv, Hv);
  pack_b_kernel<<<(Hv * Hv + T - 1) / T, T, 0, stream>>>(Whz, WhzP, Hv, Hv);
  pack_b_kernel<<<(Hv * Hv + T - 1) / T, T, 0, stream>>>(Whn, WhnP, Hv, Hv);
  pack_b_kernel<<<(Hv * Hv + T - 1) / T, T, 0, stream>>>(Wh,  WhP,  Hv, Hv);
  pack_b_kernel<<<(Hv * Ov + T - 1) / T, T, 0, stream>>>(Wo,  WoP,  Hv, Ov);

  embed_pack_kernel<<<(Bv * Sv * Ev + T - 1) / T, T, 0, stream>>>(inputs, emb, Xgru);

  // serial scan: one WGP, h double-buffered in 256 KB LDS
  gru_kernel<<<1, 512, 2 * Bv * Hv * sizeof(bf16_t), stream>>>(
      Xgru, WhrP, WhzP, WhnP, WirP, WizP, WinP, carry, bir, biz, bin_, bhn, Y,
      carry_out);

  // fused dense head across all 32768 rows
  dense_kernel<<<(Bv * Sv) / 64, 256, (size_t)Bv * Hv * sizeof(bf16_t), stream>>>(
      Y, WhP, WoP, bh, bo, logits);
}
